// MicroNetV2_5677946765987
// MI455X (gfx1250) — matrix operand order repaired and checked by reading, not compiled, not run on hardware
//
#include <hip/hip_runtime.h>
#include <math.h>

typedef __attribute__((ext_vector_type(16))) _Float16 v16h;
typedef __attribute__((ext_vector_type(8)))  _Float16 h8;
typedef __attribute__((ext_vector_type(8)))  float    v8f;

#define BN_EPS 1e-5f

__device__ __forceinline__ v8f wmma_f16(v16h a, v16h b, v8f c) {
  return __builtin_amdgcn_wmma_f32_16x16x32_f16(false, a, false, b, (short)0, c, false, false);
}
union HV { v16h v; h8 h[2]; };
// A operand: lane holds row m=lane&15; halves at [k0+ah .. +7] and [k0+ah+16 .. +23]
__device__ __forceinline__ v16h load_a16(const _Float16* __restrict__ p) {
  HV u; u.h[0] = *(const h8*)(p); u.h[1] = *(const h8*)(p + 16); return u.v;
}
// B operand: lane holds col n=lane&15; halves at [k0+bh .. +7] and [k0+bh+16 .. +23]
__device__ __forceinline__ v16h load_b16(const _Float16* __restrict__ p) {
  HV u; u.h[0] = *(const h8*)(p); u.h[1] = *(const h8*)(p + 16); return u.v;
}
__device__ __forceinline__ float gelu_exact(float x) {
  return 0.5f * x * (1.0f + erff(x * 0.70710678118654752f));
}

// ---------------- f32 -> f16 convert ----------------
__global__ void cvt_f16_kernel(const float* __restrict__ src, _Float16* __restrict__ dst, int n) {
  int i = blockIdx.x * blockDim.x + threadIdx.x;
  if (i < n) dst[i] = (_Float16)src[i];
}
// ---------------- k_w (co,ci,3,3) -> (tap, co, ci) f16 ----------------
__global__ void kw_repack_kernel(const float* __restrict__ kw, _Float16* __restrict__ kwr) {
  int i = blockIdx.x * blockDim.x + threadIdx.x;
  if (i >= 128 * 128 * 9) return;
  int tap = i % 9, ci = (i / 9) & 127, co = i / (9 * 128);
  kwr[(tap * 128 + co) * 128 + ci] = (_Float16)kw[i];
}

// ---------------- conv0 4x4 s4 + BN + GELU -> h (b,c,p) f32 ----------------
__global__ void conv0_kernel(const float* __restrict__ x, const float* __restrict__ w,
                             const float* __restrict__ bias, const float* __restrict__ g,
                             const float* __restrict__ bb, const float* __restrict__ mean,
                             const float* __restrict__ var, float* __restrict__ h) {
  int i = blockIdx.x * blockDim.x + threadIdx.x;
  if (i >= 16 * 128 * 8 * 128) return;
  int wo = i & 127, ho = (i >> 7) & 7, co = (i >> 10) & 127, b = i >> 17;
  float acc = bias[co];
  for (int ci = 0; ci < 3; ++ci)
    for (int ky = 0; ky < 4; ++ky)
      for (int kx = 0; kx < 4; ++kx)
        acc += x[((b * 3 + ci) * 32 + ho * 4 + ky) * 512 + wo * 4 + kx] *
               w[((co * 3 + ci) * 4 + ky) * 4 + kx];
  float y = (acc - mean[co]) * rsqrtf(var[co] + BN_EPS) * g[co] + bb[co];
  h[i] = gelu_exact(y);
}

// ---- depthwise 3x3 (last block) + BN + GELU + residual -> tT f16, layout (b,p,c) ----
__global__ void dw_kernel(const float* __restrict__ h, const float* __restrict__ w,
                          const float* __restrict__ bias, const float* __restrict__ g,
                          const float* __restrict__ bb, const float* __restrict__ mean,
                          const float* __restrict__ var, _Float16* __restrict__ tT) {
  int i = blockIdx.x * blockDim.x + threadIdx.x;
  if (i >= 16 * 128 * 8 * 128) return;
  int c = i & 127, p = (i >> 7) & 1023, b = i >> 17;
  int wo = p & 127, ho = p >> 7;
  float acc = bias[c];
  for (int dy = 0; dy < 3; ++dy)
    for (int dx = 0; dx < 3; ++dx) {
      int hh = ho + dy - 1, ww = wo + dx - 1;
      if (hh >= 0 && hh < 8 && ww >= 0 && ww < 128)
        acc += h[((b * 128 + c) * 8 + hh) * 128 + ww] * w[c * 9 + dy * 3 + dx];
    }
  float y = (acc - mean[c]) * rsqrtf(var[c] + BN_EPS) * g[c] + bb[c];
  tT[i] = (_Float16)(h[((b * 128 + c) * 8 + ho) * 128 + wo] + gelu_exact(y));
}

// ---- pointwise 1x1 WMMA + BN + GELU -> feat f32 (b,c,p), featT f16 (b,p,c),
//      featv f16 ((b*128+w), c*8+hf) ----
__global__ void pw_wmma_kernel(const _Float16* __restrict__ tT, const _Float16* __restrict__ wf,
                               const float* __restrict__ bias, const float* __restrict__ g,
                               const float* __restrict__ bb, const float* __restrict__ mean,
                               const float* __restrict__ var, float* __restrict__ feat,
                               _Float16* __restrict__ featT, _Float16* __restrict__ featv) {
  int lane = threadIdx.x & 31, wave = threadIdx.x >> 5;
  int p0 = (blockIdx.x * 4 + wave) * 64;
  int co0 = blockIdx.y * 16, b = blockIdx.z;
  int l15 = lane & 15;
  int ah = (lane >= 16) ? 8 : 0, bh = (lane >= 16) ? 8 : 0;
  const _Float16* arow = wf + (co0 + l15) * 128;
  v8f acc[4] = {};
#pragma unroll
  for (int k0 = 0; k0 < 128; k0 += 32) {
    v16h A = load_a16(arow + k0 + ah);
#pragma unroll
    for (int nt = 0; nt < 4; ++nt) {
      int bn = p0 + nt * 16 + l15;
      v16h B = load_b16(tT + (b * 1024 + bn) * 128 + k0 + bh);
      acc[nt] = wmma_f16(A, B, acc[nt]);
    }
  }
  for (int nt = 0; nt < 4; ++nt)
    for (int r = 0; r < 8; ++r) {
      int mm = co0 + r + ah;
      int nn = p0 + nt * 16 + l15;
      float xv = acc[nt][r] + bias[mm];
      float y = (xv - mean[mm]) * rsqrtf(var[mm] + BN_EPS) * g[mm] + bb[mm];
      float ge = gelu_exact(y);
      feat[(b * 128 + mm) * 1024 + nn] = ge;
      featT[(b * 1024 + nn) * 128 + mm] = (_Float16)ge;
      featv[(b * 128 + (nn & 127)) * 1024 + mm * 8 + (nn >> 7)] = (_Float16)ge;
    }
}

// ---- encoder input projection: xp[(b*128+t)*192+g] = featv row . wih row + bih ----
__global__ void encxp_wmma_kernel(const _Float16* __restrict__ featv,
                                  const _Float16* __restrict__ wihf,
                                  const float* __restrict__ bih, float* __restrict__ xp) {
  int lane = threadIdx.x & 31, wave = threadIdx.x >> 5;
  int n0 = (blockIdx.x * 4 + wave) * 64;
  int g0 = blockIdx.y * 16;
  int l15 = lane & 15;
  int ah = (lane >= 16) ? 8 : 0, bh = (lane >= 16) ? 8 : 0;
  const _Float16* arow = wihf + (g0 + l15) * 1024;
  v8f acc[4] = {};
#pragma unroll 4
  for (int k0 = 0; k0 < 1024; k0 += 32) {
    v16h A = load_a16(arow + k0 + ah);
#pragma unroll
    for (int nt = 0; nt < 4; ++nt) {
      v16h B = load_b16(featv + (n0 + nt * 16 + l15) * 1024 + k0 + bh);
      acc[nt] = wmma_f16(A, B, acc[nt]);
    }
  }
  for (int nt = 0; nt < 4; ++nt)
    for (int r = 0; r < 8; ++r) {
      int gg = g0 + r + ah;
      int nn = n0 + nt * 16 + l15;
      xp[nn * 192 + gg] = acc[nt][r] + bih[gg];
    }
}

// ---- encoder GRU scan (128 steps), block per batch, only final h kept ----
__global__ void enc_scan_kernel(const float* __restrict__ xp, const float* __restrict__ whh,
                                const float* __restrict__ bhh, float* __restrict__ out_enc) {
  int b = blockIdx.x;
  int g = threadIdx.x;  // 0..191
  __shared__ float h[64];
  __shared__ float hp[192];
  if (g < 64) h[g] = 0.f;
  __syncthreads();
  for (int t = 0; t < 128; ++t) {
    float acc = bhh[g];
    for (int j = 0; j < 64; ++j) acc += h[j] * whh[g * 64 + j];
    hp[g] = acc;
    __syncthreads();
    if (g < 64) {
      const float* xt = xp + (b * 128 + t) * 192;
      float r = 1.f / (1.f + expf(-(xt[g] + hp[g])));
      float z = 1.f / (1.f + expf(-(xt[64 + g] + hp[64 + g])));
      float n = tanhf(xt[128 + g] + r * hp[128 + g]);
      h[g] = (1.f - z) * n + z * h[g];
    }
    __syncthreads();
  }
  if (g < 64) out_enc[b * 64 + g] = h[g];
}

// ---- decoder input: [out_enc ; emb[targets]] ----
__global__ void indec_kernel(const float* __restrict__ out_enc, const float* __restrict__ emb,
                             const int* __restrict__ targets, float* __restrict__ in_dec) {
  int i = blockIdx.x * blockDim.x + threadIdx.x;
  if (i >= 16 * 41 * 64) return;
  int j = i & 63, t = (i >> 6) % 41, b = i / (41 * 64);
  float v;
  if (t == 0) v = out_enc[b * 64 + j];
  else        v = emb[targets[b * 40 + (t - 1)] * 64 + j];
  in_dec[i] = v;
}

// ---- decoder GRU scan (41 steps) with fused input projection; stores all y ----
__global__ void dec_scan_kernel(const float* __restrict__ in_dec, const float* __restrict__ wih,
                                const float* __restrict__ bih, const float* __restrict__ whh,
                                const float* __restrict__ bhh, float* __restrict__ y) {
  int b = blockIdx.x;
  int g = threadIdx.x;  // 0..191
  __shared__ float h[64];
  __shared__ float hp[192];
  __shared__ float xt[192];
  if (g < 64) h[g] = 0.f;
  __syncthreads();
  for (int t = 0; t < 41; ++t) {
    const float* xr = in_dec + (b * 41 + t) * 64;
    float ax = bih[g], ah = bhh[g];
    for (int j = 0; j < 64; ++j) {
      ax += xr[j] * wih[g * 64 + j];
      ah += h[j] * whh[g * 64 + j];
    }
    xt[g] = ax; hp[g] = ah;
    __syncthreads();
    if (g < 64) {
      float r = 1.f / (1.f + expf(-(xt[g] + hp[g])));
      float z = 1.f / (1.f + expf(-(xt[64 + g] + hp[64 + g])));
      float n = tanhf(xt[128 + g] + r * hp[128 + g]);
      float hn = (1.f - z) * n + z * h[g];
      h[g] = hn;
      y[(b * 41 + t) * 64 + g] = hn;
    }
    __syncthreads();
  }
}

// ---- q = y @ q_w^T + q_b ----
__global__ void q_kernel(const float* __restrict__ y, const float* __restrict__ qw,
                         const float* __restrict__ qb, float* __restrict__ q) {
  int i = blockIdx.x * blockDim.x + threadIdx.x;
  if (i >= 16 * 41 * 128) return;
  int co = i & 127, bt = i >> 7;
  float acc = qb[co];
  const float* yr = y + bt * 64;
  for (int j = 0; j < 64; ++j) acc += yr[j] * qw[co * 64 + j];
  q[i] = acc;
}

// ---- k conv 3x3 pad1 via 9 shifted WMMA GEMMs -> kk (b,c,p) f32 ----
__global__ void kconv_wmma_kernel(const _Float16* __restrict__ featT,
                                  const _Float16* __restrict__ kwr,
                                  const float* __restrict__ kb, float* __restrict__ kk) {
  int lane = threadIdx.x & 31, wave = threadIdx.x >> 5;
  int p0 = (blockIdx.x * 4 + wave) * 64;
  int co0 = blockIdx.y * 16, b = blockIdx.z;
  int l15 = lane & 15;
  int ah = (lane >= 16) ? 8 : 0, bh = (lane >= 16) ? 8 : 0;
  v8f acc[4] = {};
  for (int dy = 0; dy < 3; ++dy)
    for (int dx = 0; dx < 3; ++dx) {
      const _Float16* arow = kwr + ((dy * 3 + dx) * 128 + co0 + l15) * 128;
      int dp = (dy - 1) * 128 + (dx - 1);
#pragma unroll
      for (int k0 = 0; k0 < 128; k0 += 32) {
        v16h A = load_a16(arow + k0 + ah);
#pragma unroll
        for (int nt = 0; nt < 4; ++nt) {
          int n = p0 + nt * 16 + l15;
          int hs = (n >> 7) + dy - 1, wsft = (n & 127) + dx - 1;
          v16h B;
          if (hs >= 0 && hs < 8 && wsft >= 0 && wsft < 128) {
            B = load_b16(featT + (b * 1024 + n + dp) * 128 + k0 + bh);
          } else {
            v16h z = {}; B = z;
          }
          acc[nt] = wmma_f16(A, B, acc[nt]);
        }
      }
    }
  for (int nt = 0; nt < 4; ++nt)
    for (int r = 0; r < 8; ++r) {
      int mm = co0 + r + ah;
      kk[(b * 128 + mm) * 1024 + p0 + nt * 16 + l15] = acc[nt][r] + kb[mm];
    }
}

// ---- fused attention energy: e = sum_c tanh(kk + q)*e_w + e_b ----
__global__ void attn_e_kernel(const float* __restrict__ kk, const float* __restrict__ q,
                              const float* __restrict__ ew, const float* __restrict__ eb,
                              float* __restrict__ e) {
  int i = blockIdx.x * blockDim.x + threadIdx.x;
  if (i >= 16 * 41 * 1024) return;
  int p = i & 1023, t = (i >> 10) % 41, b = i / (41 * 1024);
  const float* kr = kk + b * 128 * 1024 + p;
  const float* qr = q + (b * 41 + t) * 128;
  float acc = eb[0];
  for (int c = 0; c < 128; ++c) acc += tanhf(kr[c * 1024] + qr[c]) * ew[c];
  e[i] = acc;
}

// ---- softmax over 1024, in place, one block per (b,t) row ----
__global__ void softmax_kernel(float* __restrict__ e) {
  float* p = e + blockIdx.x * 1024;
  __shared__ float red[256];
  int t = threadIdx.x;
  float mx = -3.0e38f;
  for (int i = t; i < 1024; i += 256) mx = fmaxf(mx, p[i]);
  red[t] = mx; __syncthreads();
  for (int s = 128; s > 0; s >>= 1) { if (t < s) red[t] = fmaxf(red[t], red[t + s]); __syncthreads(); }
  mx = red[0]; __syncthreads();
  float sm = 0.f;
  for (int i = t; i < 1024; i += 256) { float ex = expf(p[i] - mx); p[i] = ex; sm += ex; }
  red[t] = sm; __syncthreads();
  for (int s = 128; s > 0; s >>= 1) { if (t < s) red[t] += red[t + s]; __syncthreads(); }
  float inv = 1.f / red[0];
  for (int i = t; i < 1024; i += 256) p[i] *= inv;
}

// ---- attn_feat[b,t,c] = sum_p feat[b,c,p]*a[b,t,p]; emits f16 copy for FC ----
__global__ void attn_kernel(const float* __restrict__ feat, const float* __restrict__ a,
                            float* __restrict__ attn, _Float16* __restrict__ attnf) {
  int i = blockIdx.x * blockDim.x + threadIdx.x;
  if (i >= 16 * 41 * 128) return;
  int c = i & 127, t = (i >> 7) % 41, b = i / (41 * 128);
  const float* fr = feat + (b * 128 + c) * 1024;
  const float* ar = a + (b * 41 + t) * 1024;
  float acc = 0.f;
  for (int p = 0; p < 1024; ++p) acc += fr[p] * ar[p];
  attn[i] = acc;
  attnf[i] = (_Float16)acc;
}

// ---- FC head: out = attn @ fc_w^T + fc_b  (656 x 6625, K=128) via WMMA ----
__global__ void fc_wmma_kernel(const _Float16* __restrict__ attnf,
                               const _Float16* __restrict__ fcwf,
                               const float* __restrict__ fcb, float* __restrict__ out) {
  int lane = threadIdx.x & 31, wave = threadIdx.x >> 5;
  int cls0 = (blockIdx.x * 4 + wave) * 64;
  int row0 = blockIdx.y * 16;
  int l15 = lane & 15;
  int ah = (lane >= 16) ? 8 : 0, bh = (lane >= 16) ? 8 : 0;
  const _Float16* arow = attnf + (row0 + l15) * 128;
  v8f acc[4] = {};
#pragma unroll
  for (int k0 = 0; k0 < 128; k0 += 32) {
    v16h A = load_a16(arow + k0 + ah);
#pragma unroll
    for (int nt = 0; nt < 4; ++nt) {
      int n = cls0 + nt * 16 + l15;  // may read past 6625; stores are guarded
      v16h B = load_b16(fcwf + n * 128 + k0 + bh);
      acc[nt] = wmma_f16(A, B, acc[nt]);
    }
  }
  for (int nt = 0; nt < 4; ++nt)
    for (int r = 0; r < 8; ++r) {
      int mm = row0 + r + ah;
      int nn = cls0 + nt * 16 + l15;
      if (nn < 6625) out[mm * 6625 + nn] = acc[nt][r] + fcb[nn];
    }
}

extern "C" void kernel_launch(void* const* d_in, const int* in_sizes, int n_in,
                              void* d_out, int out_size, void* d_ws, size_t ws_size,
                              hipStream_t stream) {
  const float* x        = (const float*)d_in[0];
  const int*   targets  = (const int*)d_in[1];
  const float* conv0_w  = (const float*)d_in[2];
  const float* conv0_b  = (const float*)d_in[3];
  const float* bn0_g    = (const float*)d_in[4];
  const float* bn0_b    = (const float*)d_in[5];
  const float* bn0_m    = (const float*)d_in[6];
  const float* bn0_v    = (const float*)d_in[7];
  // block index 1 (DEPTH-1): block 0's output is dead code in the reference
  const float* dw_w   = (const float*)d_in[8]  + 128 * 9;
  const float* dw_b   = (const float*)d_in[9]  + 128;
  const float* bn1_g  = (const float*)d_in[10] + 128;
  const float* bn1_b  = (const float*)d_in[11] + 128;
  const float* bn1_m  = (const float*)d_in[12] + 128;
  const float* bn1_v  = (const float*)d_in[13] + 128;
  const float* pw_w   = (const float*)d_in[14] + 128 * 128;
  const float* pw_b   = (const float*)d_in[15] + 128;
  const float* bn2_g  = (const float*)d_in[16] + 128;
  const float* bn2_b  = (const float*)d_in[17] + 128;
  const float* bn2_m  = (const float*)d_in[18] + 128;
  const float* bn2_v  = (const float*)d_in[19] + 128;
  const float* enc_wih = (const float*)d_in[20];
  const float* enc_whh = (const float*)d_in[21];
  const float* enc_bih = (const float*)d_in[22];
  const float* enc_bhh = (const float*)d_in[23];
  const float* dec_wih = (const float*)d_in[24];
  const float* dec_whh = (const float*)d_in[25];
  const float* dec_bih = (const float*)d_in[26];
  const float* dec_bhh = (const float*)d_in[27];
  const float* emb     = (const float*)d_in[28];
  const float* q_w     = (const float*)d_in[29];
  const float* q_b     = (const float*)d_in[30];
  const float* k_w     = (const float*)d_in[31];
  const float* k_b     = (const float*)d_in[32];
  const float* e_w     = (const float*)d_in[33];
  const float* e_b     = (const float*)d_in[34];
  const float* fc_w    = (const float*)d_in[35];
  const float* fc_b    = (const float*)d_in[36];
  float* out = (float*)d_out;

  char* ws = (char*)d_ws;
  size_t off = 0;
  auto alloc = [&](size_t bytes) -> void* {
    void* p = (void*)(ws + off);
    off = (off + bytes + 255) & ~(size_t)255;
    return p;
  };
  const int NPIX = 16 * 128 * 8 * 128;  // 2,097,152
  float*    h        = (float*)alloc((size_t)NPIX * 4);
  _Float16* tT_f16   = (_Float16*)alloc((size_t)NPIX * 2);
  float*    feat     = (float*)alloc((size_t)NPIX * 4);
  _Float16* featT    = (_Float16*)alloc((size_t)NPIX * 2);
  _Float16* featv    = (_Float16*)alloc((size_t)NPIX * 2);
  _Float16* pw_w_f16 = (_Float16*)alloc(16384 * 2);
  _Float16* wih_f16  = (_Float16*)alloc(192 * 1024 * 2);
  _Float16* kwr_f16  = (_Float16*)alloc(128 * 128 * 9 * 2);
  _Float16* fcw_f16  = (_Float16*)alloc((size_t)6625 * 128 * 2);  // OOB reads fall into xp: safe
  float*    xp       = (float*)alloc(2048 * 192 * 4);
  float*    out_enc  = (float*)alloc(16 * 64 * 4);
  float*    in_dec   = (float*)alloc(16 * 41 * 64 * 4);
  float*    ybuf     = (float*)alloc(16 * 41 * 64 * 4);
  float*    qbuf     = (float*)alloc(16 * 41 * 128 * 4);
  float*    kkbuf    = (float*)alloc((size_t)NPIX * 4);
  float*    ebuf     = (float*)alloc(16 * 41 * 1024 * 4);
  float*    attn     = (float*)alloc(16 * 41 * 128 * 4);
  _Float16* attn_f16 = (_Float16*)alloc(16 * 41 * 128 * 2);

  // weight conversions / repacks
  cvt_f16_kernel<<<(16384 + 255) / 256, 256, 0, stream>>>(pw_w, pw_w_f16, 16384);
  cvt_f16_kernel<<<(192 * 1024 + 255) / 256, 256, 0, stream>>>(enc_wih, wih_f16, 192 * 1024);
  kw_repack_kernel<<<(128 * 128 * 9 + 255) / 256, 256, 0, stream>>>(k_w, kwr_f16);
  cvt_f16_kernel<<<(6625 * 128 + 255) / 256, 256, 0, stream>>>(fc_w, fcw_f16, 6625 * 128);

  // stem + last block
  conv0_kernel<<<NPIX / 256, 256, 0, stream>>>(x, conv0_w, conv0_b, bn0_g, bn0_b, bn0_m, bn0_v, h);
  dw_kernel<<<NPIX / 256, 256, 0, stream>>>(h, dw_w, dw_b, bn1_g, bn1_b, bn1_m, bn1_v, tT_f16);
  pw_wmma_kernel<<<dim3(4, 8, 16), 128, 0, stream>>>(tT_f16, pw_w_f16, pw_b, bn2_g, bn2_b,
                                                     bn2_m, bn2_v, feat, featT, featv);

  // encoder (WMMA projection + per-batch scan; only final h needed)
  encxp_wmma_kernel<<<dim3(8, 12), 128, 0, stream>>>(featv, wih_f16, enc_bih, xp);
  enc_scan_kernel<<<16, 192, 0, stream>>>(xp, enc_whh, enc_bhh, out_enc);

  // decoder
  indec_kernel<<<(16 * 41 * 64 + 255) / 256, 256, 0, stream>>>(out_enc, emb, targets, in_dec);
  dec_scan_kernel<<<16, 192, 0, stream>>>(in_dec, dec_wih, dec_bih, dec_whh, dec_bhh, ybuf);
  q_kernel<<<(16 * 41 * 128 + 255) / 256, 256, 0, stream>>>(ybuf, q_w, q_b, qbuf);

  // attention
  kconv_wmma_kernel<<<dim3(4, 8, 16), 128, 0, stream>>>(featT, kwr_f16, k_b, kkbuf);
  attn_e_kernel<<<(16 * 41 * 1024 + 255) / 256, 256, 0, stream>>>(kkbuf, qbuf, e_w, e_b, ebuf);
  softmax_kernel<<<16 * 41, 256, 0, stream>>>(ebuf);
  attn_kernel<<<(16 * 41 * 128 + 255) / 256, 256, 0, stream>>>(feat, ebuf, attn, attn_f16);

  // classifier head (656 rows = 41 exact M-tiles)
  fc_wmma_kernel<<<dim3(26, 41), 128, 0, stream>>>(attn_f16, fcw_f16, fc_b, out);
}
